// LSTM_27281632264361
// MI455X (gfx1250) — compile-verified, matrix operand order checked by reading, not run on hardware
//
#include <hip/hip_runtime.h>
#include <hip/hip_bf16.h>
#include <stdint.h>

#define B_   2048
#define T_   512
#define H_   128
#define I_   64
#define G_   512     // 4*H
#define MHU  1.5f

typedef __attribute__((ext_vector_type(16))) __bf16 v16bf;
typedef __attribute__((ext_vector_type(8)))  float  v8f;
typedef __attribute__((ext_vector_type(4)))  unsigned int u32x4;
typedef __attribute__((ext_vector_type(8)))  int i32x8;
typedef __attribute__((ext_vector_type(4)))  int i32x4;

#if defined(__has_builtin)
#if __has_builtin(__builtin_amdgcn_tensor_load_to_lds) && __has_builtin(__builtin_amdgcn_s_wait_tensorcnt)
#define USE_TDM 1
#endif
#endif

struct __align__(16) U4 { unsigned int x, y, z, w; };

union Frag16 {
    __bf16 e[16];
    v16bf  v;
    U4     q[2];
};

static __device__ __forceinline__ __bf16 f2bf(float x) {
    union { float f; uint32_t u; } a; a.f = x;
    uint32_t r = (a.u + 0x7fffu + ((a.u >> 16) & 1u)) >> 16;   // round-to-nearest-even
    union { uint16_t s; __bf16 b; } o; o.s = (uint16_t)r;
    return o.b;
}

static __device__ __forceinline__ float sigf(float x) {
    return 1.0f / (1.0f + __expf(-x));
}

// tanh via v_exp_f32: tanh(x) = sign(x) * (1 - 2/(e^{2|x|}+1)); saturates cleanly.
static __device__ __forceinline__ float tanh_fast(float x) {
    float ax = fabsf(x);
    float e  = __expf(2.0f * ax);
    float r  = 1.0f - 2.0f / (e + 1.0f);
    return copysignf(r, x);
}

// ---------------------------------------------------------------------------
// Kernel 1: convert U (512x128 f32, row-major [n][k]) to bf16 in workspace.
// ---------------------------------------------------------------------------
__global__ void u_to_bf16_kernel(const float* __restrict__ U, __bf16* __restrict__ Ub) {
    int i = blockIdx.x * blockDim.x + threadIdx.x;
    if (i < G_ * H_) Ub[i] = f2bf(U[i]);
}

// ---------------------------------------------------------------------------
// Kernel 2: Wd = data @ W.T + bW  (M=2048, K=64, N=512) via bf16 WMMA.
// grid (128, 4), block 256 (8 waves). Wave w of block (x,y) does n-tile y*8+w.
// ---------------------------------------------------------------------------
__global__ __launch_bounds__(256)
void wd_gemm_kernel(const float* __restrict__ data, const float* __restrict__ W,
                    const float* __restrict__ bW, float* __restrict__ Wd) {
    const int lane = threadIdx.x & 31;
    const int wv   = threadIdx.x >> 5;
    const int b0   = blockIdx.x * 16;
    const int n0   = (blockIdx.y * 8 + wv) * 16;
    const int l15  = lane & 15;
    const int kh   = (lane >> 4) * 16;
    const int hi8  = (lane >> 4) * 8;

    float bias = bW[n0 + l15];
    v8f acc;
#pragma unroll
    for (int r = 0; r < 8; ++r) acc[r] = bias;

#pragma unroll
    for (int kc = 0; kc < 2; ++kc) {
        Frag16 a, b;
        const float* ap = data + (b0 + l15) * I_ + kc * 32 + kh;
        const float* bp = W    + (n0 + l15) * I_ + kc * 32 + kh;
#pragma unroll
        for (int j = 0; j < 16; ++j) { a.e[j] = f2bf(ap[j]); b.e[j] = f2bf(bp[j]); }
        acc = __builtin_amdgcn_wmma_f32_16x16x32_bf16(false, a.v, false, b.v,
                                                      (short)0, acc, false, false);
    }
#pragma unroll
    for (int r = 0; r < 8; ++r)
        Wd[(b0 + hi8 + r) * G_ + n0 + l15] = acc[r];
}

// ---------------------------------------------------------------------------
// Kernel 3: the recurrence. 128 workgroups x 16 batch rows, 256 threads
// (8 waves). U lives in LDS as bf16 (128 KB, staged by the Tensor Data Mover
// when available). Wave w computes gate tiles {w, w+8, w+16, w+24} == i,f,g,o
// for exactly its own h columns, so gates stay in registers.
// ---------------------------------------------------------------------------
__global__ __launch_bounds__(256)
void lstm_kernel(const float* __restrict__ Wd, const __bf16* __restrict__ Ub,
                 const float* __restrict__ h0, const float* __restrict__ c0,
                 const float* __restrict__ tau, const int* __restrict__ bsz,
                 float* __restrict__ out) {
    extern __shared__ char smem[];
    __bf16* u_lds = (__bf16*)smem;                       // 512*128 bf16 = 128 KB
    __bf16* h_lds = (__bf16*)(smem + G_ * H_ * 2);       // 16*128 bf16 = 4 KB

    const int tid  = threadIdx.x;
    const int lane = tid & 31;
    const int wv   = tid >> 5;
    const int b0   = blockIdx.x * 16;
    const int l15  = lane & 15;
    const int hi8  = (lane >> 4) * 8;
    const int kh   = (lane >> 4) * 16;

    // ---- Stage U into LDS ------------------------------------------------
#if USE_TDM
    if (wv == 0) {
        // D# for a 2-D bf16 tensor: 512 rows x 128 cols, contiguous.
        const uint32_t lds_base = (uint32_t)(uintptr_t)u_lds;
        const uint64_t ga = (uint64_t)(uintptr_t)Ub;
        u32x4 g0;
        g0[0] = 1u;                                      // count=1, user mode
        g0[1] = lds_base;                                // lds_addr
        g0[2] = (uint32_t)ga;                            // global_addr[31:0]
        g0[3] = (uint32_t)(ga >> 32) | (2u << 30);       // global_addr[56:32] | type=2
        i32x8 g1;
        g1[0] = (int)(1u << 16);                         // data_size = 2 bytes
        g1[1] = (int)(128u << 16);                       // tensor_dim0 = 128
        g1[2] = (int)(512u << 16);                       // tensor_dim1 = 512
        g1[3] = (int)(128u << 16);                       // tile_dim0 = 128
        g1[4] = 512;                                     // tile_dim1 = 512
        g1[5] = 128;                                     // tensor_dim0_stride = 128
        g1[6] = 0;
        g1[7] = 0;
        i32x4 z4 = {0, 0, 0, 0};
#if defined(__clang_major__) && (__clang_major__ >= 23)
        i32x8 z8 = {0, 0, 0, 0, 0, 0, 0, 0};
        __builtin_amdgcn_tensor_load_to_lds(g0, g1, z4, z4, z8, 0);
#else
        __builtin_amdgcn_tensor_load_to_lds(g0, g1, z4, z4, 0);
#endif
        __builtin_amdgcn_s_wait_tensorcnt(0);
    }
#else
    {
        const U4* src = (const U4*)Ub;
        U4* dst = (U4*)u_lds;
        const int total = G_ * H_ / 8;
        for (int i = tid; i < total; i += 256) dst[i] = src[i];
    }
#endif

    // Per-lane recurrent state: h,c for column n_h, rows hi8+0..7.
    const int n_h = wv * 16 + l15;
    float h[8], c[8];
#pragma unroll
    for (int r = 0; r < 8; ++r) {
        int b = b0 + hi8 + r;
        h[r] = h0[b * H_ + n_h];
        c[r] = c0[b * H_ + n_h];
    }

    // Resident Wd tiles for this wave (C-layout).
    v8f wd[4];
#pragma unroll
    for (int g = 0; g < 4; ++g) {
        int n = (wv + g * 8) * 16 + l15;
#pragma unroll
        for (int r = 0; r < 8; ++r)
            wd[g][r] = Wd[(b0 + hi8 + r) * G_ + n];
    }

    const float tau_s = tau[0];
    const float k0 = tau_s * MHU;         // tau * 1.5
    const float k1 = tau_s / MHU;         // tau / 1.5

    __syncthreads();                      // U staged & visible

    for (int t = 0; t < T_; ++t) {
        const int bt = bsz[t];

        // mhu rotation on h[:, :2] (wave 0 owns columns 0..15).
        if (t != 0 && wv == 0) {
#pragma unroll
            for (int r = 0; r < 8; ++r) {
                float part = __shfl_xor(h[r], 1, 32);
                if (l15 == 0)      h[r] = h[r] + k0 * h[r] + k1 * part;
                else if (l15 == 1) h[r] = h[r] - k0 * part;
            }
        }

        // Publish h tile as bf16 (shared A operand).
#pragma unroll
        for (int r = 0; r < 8; ++r)
            h_lds[(hi8 + r) * H_ + n_h] = f2bf(h[r]);

        __syncthreads();

        // A fragments: lane m = l15, K-chunk kc, half kh -> 16 contiguous bf16.
        Frag16 afr[4];
#pragma unroll
        for (int kc = 0; kc < 4; ++kc) {
            const U4* p = (const U4*)(h_lds + l15 * H_ + kc * 32 + kh);
            afr[kc].q[0] = p[0];
            afr[kc].q[1] = p[1];
        }

        v8f acc[4];
#pragma unroll
        for (int g = 0; g < 4; ++g) acc[g] = wd[g];

#pragma unroll
        for (int kc = 0; kc < 4; ++kc) {
#pragma unroll
            for (int g = 0; g < 4; ++g) {
                const int n = (wv + g * 8) * 16 + l15;
                Frag16 bfr;
                const U4* p = (const U4*)(u_lds + n * H_ + kc * 32 + kh);
                bfr.q[0] = p[0];
                bfr.q[1] = p[1];
                acc[g] = __builtin_amdgcn_wmma_f32_16x16x32_bf16(
                    false, afr[kc].v, false, bfr.v, (short)0, acc[g], false, false);
            }
        }

        // Elementwise: acc[0]=i, acc[1]=f, acc[2]=g, acc[3]=o.
#pragma unroll
        for (int r = 0; r < 8; ++r) {
            int b = b0 + hi8 + r;
            float cn = sigf(acc[1][r]) * c[r] + sigf(acc[0][r]) * tanh_fast(acc[2][r]);
            float hn = sigf(acc[3][r]) * tanh_fast(cn);
            float ov = 0.0f;
            if (b < bt) { c[r] = cn; h[r] = hn; ov = hn; }
            out[((size_t)b * T_ + t) * H_ + n_h] = ov;
        }

        __syncthreads();                  // h_lds reads done before next rewrite
    }

    // Final h, c appended after outputs.
    float* hout = out + (size_t)B_ * T_ * H_;
    float* cout = hout + (size_t)B_ * H_;
#pragma unroll
    for (int r = 0; r < 8; ++r) {
        int b = b0 + hi8 + r;
        hout[b * H_ + n_h] = h[r];
        cout[b * H_ + n_h] = c[r];
    }
}

// ---------------------------------------------------------------------------
extern "C" void kernel_launch(void* const* d_in, const int* in_sizes, int n_in,
                              void* d_out, int out_size, void* d_ws, size_t ws_size,
                              hipStream_t stream) {
    const float* data = (const float*)d_in[0];
    const float* W    = (const float*)d_in[1];
    const float* bW   = (const float*)d_in[2];
    const float* U    = (const float*)d_in[3];
    const float* h0   = (const float*)d_in[4];
    const float* c0   = (const float*)d_in[5];
    const float* tau  = (const float*)d_in[6];
    const int*   bsz  = (const int*)d_in[7];
    float* out = (float*)d_out;

    __bf16* Ub = (__bf16*)d_ws;                                   // 128 KB
    float*  Wd = (float*)((char*)d_ws + (size_t)G_ * H_ * 2);     // 4 MB

    u_to_bf16_kernel<<<dim3((G_ * H_ + 255) / 256), dim3(256), 0, stream>>>(U, Ub);
    wd_gemm_kernel<<<dim3(B_ / 16, 4), dim3(256), 0, stream>>>(data, W, bW, Wd);

    size_t smem = (size_t)(G_ * H_ + 16 * H_) * 2;                // 132 KB LDS
    lstm_kernel<<<dim3(B_ / 16), dim3(256), smem, stream>>>(Wd, Ub, h0, c0, tau, bsz, out);
}
